// GMMLoss_pca_18339510354133
// MI455X (gfx1250) — matrix operand order repaired and checked by reading, not compiled, not run on hardware
//
#include <hip/hip_runtime.h>
#include <cstdint>
#include <cstddef>

// ---------------------------------------------------------------------------
// GMM KL loss with PCA front-end, CDNA5 (gfx1250) WMMA bf16 implementation.
//   B=8192 samples, F=2048 features, D=256 PCA dims, L=10 labels.
// GEMMs use v_wmma_f32_16x16x32_bf16 with double-buffered LDS tiles
// (K-innermost layout: fragment = 2x ds_load_b128), ping-ponged B fragments
// (partial s_wait_dscnt), TDM staging (tensor_load_to_lds + s_wait_tensorcnt)
// for gemm_nn A tiles, and shift/mask split-K for the label scatter.
// ---------------------------------------------------------------------------

typedef __attribute__((ext_vector_type(16))) __bf16 v16bf;
typedef __attribute__((ext_vector_type(8)))  float  v8f;
typedef __attribute__((ext_vector_type(4)))  unsigned int u32x4;
typedef __attribute__((ext_vector_type(8)))  int i32x8;
typedef __attribute__((ext_vector_type(4)))  int i32x4;

constexpr int NB = 8192;   // batch
constexpr int NF = 2048;   // features
constexpr int ND = 256;    // PCA dim
constexpr int NL = 10;     // labels
constexpr int KSHIFT = 3;  // log2 split-K factor for the label scatter
constexpr int KS = 1 << KSHIFT;
constexpr float SIGMA_REG = 1.0f;

constexpr int LDK = 40;    // padded LDS row length (32 K-elems + 8 pad bf16)

#if defined(__has_builtin)
#if __has_builtin(__builtin_amdgcn_tensor_load_to_lds)
#define HAVE_TDM 1
#endif
#if __has_builtin(__builtin_amdgcn_s_wait_tensorcnt)
#define HAVE_TCNT 1
#endif
#endif

union Frag16 {                 // two ds_load_b128 -> one 16x32 bf16 fragment
  uint4 q[2];
  v16bf v;
};
union Pack8 {                  // one global_load_b128 -> 8 bf16
  uint4 q;
  __bf16 e[8];
};

#if defined(HAVE_TDM)
// 2D TDM tile load: (tile0 contiguous elems) x (tile1 rows, stride0 elems),
// bf16, LDS row padding 32+8 elems (pad_interval=16DW, pad_amount=4DW) so the
// landing layout is lds[row][LDK] — identical to the cooperative path.
__device__ __forceinline__ void tdm_load_2d(uint32_t lds_addr, const void* gptr,
                                            uint32_t td0, uint32_t td1,
                                            uint32_t tile0, uint32_t tile1,
                                            uint32_t stride0) {
  uint64_t ga = (uint64_t)(uintptr_t)gptr;
  u32x4 g0;
  g0[0] = 1u;                                              // count=1 user descriptor
  g0[1] = lds_addr;                                        // lds byte address
  g0[2] = (uint32_t)ga;                                    // global_addr[31:0]
  g0[3] = (uint32_t)((ga >> 32) & 0x01FFFFFFull) | (2u << 30);  // [56:32] | type=2
  i32x8 g1;
  g1[0] = (int)((1u << 16)      // data_size = 1 (2 bytes)
              | (1u << 20)      // pad_enable
              | (3u << 22)      // pad_interval code 3 -> 16 DWORDs (=32 bf16)
              | (3u << 25));    // pad_amount  code 3 -> 4 DWORDs  (=8 bf16)
  g1[1] = (int)((td0 & 0xFFFFu) << 16);                    // tensor_dim0[15:0]
  g1[2] = (int)((td0 >> 16) | ((td1 & 0xFFFFu) << 16));    // td0[31:16] | td1[15:0]
  g1[3] = (int)((td1 >> 16) | (tile0 << 16));              // td1[31:16] | tile_dim0
  g1[4] = (int)(tile1);                                    // tile_dim1 | tile_dim2=0
  g1[5] = (int)(stride0);                                  // tensor_dim0_stride lo32
  g1[6] = 0;
  g1[7] = 0;
  i32x4 z4 = {0, 0, 0, 0};
#if __clang_major__ >= 23
  i32x8 z8 = {0, 0, 0, 0, 0, 0, 0, 0};
  __builtin_amdgcn_tensor_load_to_lds(g0, g1, z4, z4, z8, 0);
#else
  __builtin_amdgcn_tensor_load_to_lds(g0, g1, z4, z4, 0);
#endif
}
#endif

// transpose-stage: lds[c][kk] = X[k0+kk][c0+c], coalesced b128 global reads
__device__ __forceinline__ void stage_transposed(const __bf16* __restrict__ X,
                                                 int k0, int c0, int M, int K,
                                                 const int* __restrict__ labels,
                                                 int lab, __bf16* __restrict__ lds,
                                                 int tid) {
  for (int c = tid; c < 512; c += 256) {
    const int kk = c >> 4;            // 0..31
    const int mb = (c & 15) << 3;     // 0,8,..,120
    Pack8 p;
    p.q = *(const uint4*)(X + (size_t)(k0 + kk) * M + c0 + mb);
    if (k0 + 64 < K)
      __builtin_prefetch(X + (size_t)(k0 + 64 + kk) * M + c0 + mb, 0, 1);
    const bool keep = !labels || (labels[k0 + kk] == lab);
#pragma unroll
    for (int j = 0; j < 8; ++j)
      lds[(mb + j) * LDK + kk] = keep ? p.e[j] : (__bf16)0.0f;
  }
}

// row-stage (k-contiguous source): lds[m][kk] = A[m0+m][k0+kk]
__device__ __forceinline__ void stage_rows(const __bf16* __restrict__ A,
                                           int m0, int k0, int K,
                                           __bf16* __restrict__ lds, int tid) {
  for (int c = tid; c < 512; c += 256) {
    const int mloc = c >> 2;
    const int kc   = (c & 3) << 3;
    *(uint4*)&lds[mloc * LDK + kc] =
        *(const uint4*)(A + (size_t)(m0 + mloc) * K + k0 + kc);
  }
}

// ---- stage 1: column mean + center/cast -----------------------------------

__global__ void mean_kernel(const float* __restrict__ mu, float* __restrict__ mean) {
  int col = blockIdx.x * blockDim.x + threadIdx.x;   // 0..NF-1
  float s = 0.0f;
  for (int b = 0; b < NB; ++b) s += mu[(size_t)b * NF + col];
  mean[col] = s * (1.0f / (float)NB);
}

__global__ void center_cast_kernel(const float* __restrict__ mu,
                                   const float* __restrict__ mean,
                                   __bf16* __restrict__ Xcb) {
  size_t i = (size_t)blockIdx.x * blockDim.x + threadIdx.x;  // over NB*NF
  int f = (int)(i % NF);
  Xcb[i] = (__bf16)(mu[i] - mean[f]);
}

__global__ void cast_f32_bf16_kernel(const float* __restrict__ in,
                                     __bf16* __restrict__ out, size_t n) {
  size_t i = (size_t)blockIdx.x * blockDim.x + threadIdx.x;
  if (i < n) out[i] = (__bf16)in[i];
}

// ---- WMMA GEMM: out[z] = X^T X over k-range (optional label row-mask) -----
// X: [K x M] row-major bf16. 8 waves/block; block computes a 128x128 tile.
// Double-buffered LDS, one barrier per k-step; B fragments ping-pong so the
// next ds_loads overlap the current WMMA. kshift = log2(split-K factor).

__global__ void ata_wmma_kernel(const __bf16* __restrict__ X, int K, int M,
                                const int* __restrict__ labels, int kshift,
                                float* __restrict__ out) {
  __shared__ __bf16 ldsA[2][128 * LDK];
  __shared__ __bf16 ldsB[2][128 * LDK];

  const int tid  = threadIdx.x;
  const int wave = tid >> 5;
  const int lane = tid & 31;
  const int hl   = (lane >> 4) & 1;
  const int nrow = lane & 15;
  const int m0   = blockIdx.x * 128;
  const int n0   = blockIdx.y * 128;
  const int lab  = blockIdx.z >> kshift;            // shift/mask: no int division
  const int kseg = blockIdx.z & ((1 << kshift) - 1);
  const int klen = K >> kshift;
  const int kb   = kseg * klen;
  const int ke   = kb + klen;

  v8f acc[8];
#pragma unroll
  for (int s = 0; s < 8; ++s)
#pragma unroll
    for (int r = 0; r < 8; ++r) acc[s][r] = 0.0f;

  stage_transposed(X, kb, m0, M, K, labels, lab, ldsA[0], tid);
  stage_transposed(X, kb, n0, M, K, nullptr, 0, ldsB[0], tid);
  __syncthreads();

  int cur = 0;
  for (int k0 = kb; k0 < ke; k0 += 32) {
    const int nxt = cur ^ 1;
    if (k0 + 32 < ke) {
      stage_transposed(X, k0 + 32, m0, M, K, labels, lab, ldsA[nxt], tid);
      stage_transposed(X, k0 + 32, n0, M, K, nullptr, 0, ldsB[nxt], tid);
    }

    Frag16 fa;
    const __bf16* ap = &ldsA[cur][(wave * 16 + nrow) * LDK];
    fa.q[0] = *(const uint4*)(ap + hl * 16);       // K = hl*16 + 0..7
    fa.q[1] = *(const uint4*)(ap + hl * 16 + 8);   // K = hl*16 + 8..15
    const __bf16* bBase = &ldsB[cur][nrow * LDK + hl * 16];
    Frag16 fb[2];
    fb[0].q[0] = *(const uint4*)(bBase);
    fb[0].q[1] = *(const uint4*)(bBase + 8);
#pragma unroll
    for (int s = 0; s < 8; ++s) {
      if (s < 7) {
        const __bf16* nb = bBase + (s + 1) * 16 * LDK;
        fb[(s + 1) & 1].q[0] = *(const uint4*)nb;
        fb[(s + 1) & 1].q[1] = *(const uint4*)(nb + 8);
      }
      acc[s] = __builtin_amdgcn_wmma_f32_16x16x32_bf16(
          false, fa.v, false, fb[s & 1].v, (short)0, acc[s], false, false);
    }
    __syncthreads();
    cur = nxt;
  }

  float* op = out + (size_t)blockIdx.z * (size_t)M * (size_t)M +
              (size_t)(m0 + wave * 16 + hl * 8) * M + n0 + nrow;
#pragma unroll
  for (int r = 0; r < 8; ++r) {
#pragma unroll
    for (int s = 0; s < 8; ++s) op[s * 16] = acc[s][r];
    op += M;
  }
}

// ---- WMMA GEMM: out = A (MxK row-major) * B (KxN row-major), bf16 in ------
// A tile is K-innermost already => staged by the Tensor Data Mover when
// available (descriptor pad fields reproduce the LDK=40 row layout).

__global__ void gemm_nn_wmma_kernel(const __bf16* __restrict__ A,
                                    const __bf16* __restrict__ Bm,
                                    float* __restrict__ out,
                                    int M, int K, int N) {
  __shared__ __bf16 ldsA[2][128 * LDK];
  __shared__ __bf16 ldsB[2][128 * LDK];

  const int tid  = threadIdx.x;
  const int wave = tid >> 5;
  const int lane = tid & 31;
  const int hl   = (lane >> 4) & 1;
  const int nrow = lane & 15;
  const int m0   = blockIdx.x * 128;
  const int n0   = blockIdx.y * 128;

  v8f acc[8];
#pragma unroll
  for (int s = 0; s < 8; ++s)
#pragma unroll
    for (int r = 0; r < 8; ++r) acc[s][r] = 0.0f;

#if defined(HAVE_TDM)
  if (wave == 0)
    tdm_load_2d((uint32_t)(uintptr_t)(const void*)ldsA[0],
                A + (size_t)m0 * K, (uint32_t)K, (uint32_t)M,
                32u, 128u, (uint32_t)K);
#else
  stage_rows(A, m0, 0, K, ldsA[0], tid);
#endif
  stage_transposed(Bm, 0, n0, N, K, nullptr, 0, ldsB[0], tid);
#if defined(HAVE_TDM) && defined(HAVE_TCNT)
  if (wave == 0) __builtin_amdgcn_s_wait_tensorcnt(0);
#endif
  __syncthreads();

  int cur = 0;
  for (int k0 = 0; k0 < K; k0 += 32) {
    const int nxt = cur ^ 1;
    const bool more = (k0 + 32 < K);
    if (more) {
#if defined(HAVE_TDM)
      if (wave == 0)
        tdm_load_2d((uint32_t)(uintptr_t)(const void*)ldsA[nxt],
                    A + (size_t)m0 * K + k0 + 32, (uint32_t)K, (uint32_t)M,
                    32u, 128u, (uint32_t)K);
#else
      stage_rows(A, m0, k0 + 32, K, ldsA[nxt], tid);
#endif
      stage_transposed(Bm, k0 + 32, n0, N, K, nullptr, 0, ldsB[nxt], tid);
    }

    Frag16 fa;
    const __bf16* ap = &ldsA[cur][(wave * 16 + nrow) * LDK];
    fa.q[0] = *(const uint4*)(ap + hl * 16);
    fa.q[1] = *(const uint4*)(ap + hl * 16 + 8);
    const __bf16* bBase = &ldsB[cur][nrow * LDK + hl * 16];
    Frag16 fb[2];
    fb[0].q[0] = *(const uint4*)(bBase);
    fb[0].q[1] = *(const uint4*)(bBase + 8);
#pragma unroll
    for (int s = 0; s < 8; ++s) {
      if (s < 7) {
        const __bf16* nb = bBase + (s + 1) * 16 * LDK;
        fb[(s + 1) & 1].q[0] = *(const uint4*)nb;
        fb[(s + 1) & 1].q[1] = *(const uint4*)(nb + 8);
      }
      acc[s] = __builtin_amdgcn_wmma_f32_16x16x32_bf16(
          false, fa.v, false, fb[s & 1].v, (short)0, acc[s], false, false);
    }
#if defined(HAVE_TDM) && defined(HAVE_TCNT)
    if (more && wave == 0) __builtin_amdgcn_s_wait_tensorcnt(0);
#endif
    __syncthreads();
    cur = nxt;
  }

  float* op = out + (size_t)(m0 + wave * 16 + hl * 8) * N + n0 + nrow;
#pragma unroll
  for (int r = 0; r < 8; ++r) {
#pragma unroll
    for (int s = 0; s < 8; ++s) op[s * 16] = acc[s][r];
    op += N;
  }
}

// ---- split-K partial reduce (deterministic; no float atomics) -------------

__global__ void reduce_ksplit_kernel(const float* __restrict__ part,
                                     float* __restrict__ sig) {
  size_t e = (size_t)blockIdx.x * blockDim.x + threadIdx.x;  // over NL*ND*ND
  size_t l = e / ((size_t)ND * ND);
  size_t r = e % ((size_t)ND * ND);
  float s = 0.0f;
#pragma unroll
  for (int ks = 0; ks < KS; ++ks)
    s += part[((size_t)l * KS + ks) * ND * ND + r];
  sig[e] = s;
}

// ---- power-iteration helpers ----------------------------------------------

__global__ void init_q_kernel(float* __restrict__ Q, __bf16* __restrict__ Qb) {
  size_t i = (size_t)blockIdx.x * blockDim.x + threadIdx.x;  // over NF*ND
  uint32_t f = (uint32_t)(i / ND), d = (uint32_t)(i % ND);
  uint32_t h = f * 2654435761u ^ (d * 2246822519u) ^ 0x9E3779B9u;
  h ^= h >> 16; h *= 0x85EBCA6Bu; h ^= h >> 13;
  float v = ((float)(h & 0xFFFFu) * (1.0f / 65536.0f)) - 0.5f;
  Q[i] = v; Qb[i] = (__bf16)v;
}

__global__ void colnorm_kernel(const float* __restrict__ Z,
                               float* __restrict__ Q, __bf16* __restrict__ Qb) {
  __shared__ float red[256];
  __shared__ float sinvn;
  const int d = blockIdx.x, tid = threadIdx.x;
  float s = 0.0f;
  for (int f = tid; f < NF; f += 256) {
    float v = Z[(size_t)f * ND + d]; s += v * v;
  }
  red[tid] = s; __syncthreads();
  for (int st = 128; st > 0; st >>= 1) {
    if (tid < st) red[tid] += red[tid + st];
    __syncthreads();
  }
  if (tid == 0) sinvn = rsqrtf(red[0] + 1e-20f);
  __syncthreads();
  const float inv = sinvn;
  for (int f = tid; f < NF; f += 256) {
    float v = Z[(size_t)f * ND + d] * inv;
    Q[(size_t)f * ND + d] = v; Qb[(size_t)f * ND + d] = (__bf16)v;
  }
}

// ---- label statistics ------------------------------------------------------

__global__ void counts_kernel(const int* __restrict__ lbl, float* __restrict__ counts) {
  __shared__ float sc[16];
  const int tid = threadIdx.x;
  if (tid < 16) sc[tid] = 0.0f;
  __syncthreads();
  for (int b = tid; b < NB; b += 256) atomicAdd(&sc[lbl[b]], 1.0f);
  __syncthreads();
  if (tid < NL) counts[tid] = sc[tid];
}

__global__ void mus_kernel(const int* __restrict__ lbl, const float* __restrict__ P,
                           const float* __restrict__ counts, float* __restrict__ mus) {
  const int l = blockIdx.x, d = threadIdx.x;
  float s = 0.0f;
  for (int b = 0; b < NB; ++b)
    if (lbl[b] == l) s += P[(size_t)b * ND + d];
  mus[(size_t)l * ND + d] = s / fmaxf(counts[l], 1.0f);
}

__global__ void sig_kernel(float* __restrict__ sig, const float* __restrict__ counts,
                           const float* __restrict__ mus) {
  size_t e = (size_t)blockIdx.x * blockDim.x + threadIdx.x;  // over NL*ND*ND
  int l = (int)(e / ((size_t)ND * ND));
  int rem = (int)(e % ((size_t)ND * ND));
  int i = rem / ND, j = rem % ND;
  float c = counts[l], safe = fmaxf(c, 1.0f);
  float v = (sig[e] - c * mus[(size_t)l * ND + i] * mus[(size_t)l * ND + j]) / safe;
  sig[e] = v + ((i == j) ? SIGMA_REG : 0.0f);
}

// ---- small linear algebra: Cholesky, inverse, KL --------------------------

__global__ void cholesky_kernel(const float* __restrict__ sig, float* __restrict__ chol,
                                float* __restrict__ logdet) {
  const int l = blockIdx.x, tid = threadIdx.x;
  float* Lc = chol + (size_t)l * ND * ND;
  const float* S = sig + (size_t)l * ND * ND;
  for (int e = tid; e < ND * ND; e += 256) Lc[e] = S[e];
  __syncthreads();
  for (int k = 0; k < ND; ++k) {
    if (tid == 0) Lc[k * ND + k] = sqrtf(fmaxf(Lc[k * ND + k], 1e-20f));
    __syncthreads();
    const float dk = Lc[k * ND + k];
    for (int r = k + 1 + tid; r < ND; r += 256) Lc[r * ND + k] /= dk;
    __syncthreads();
    // trailing update: thread c owns column c (division-free mapping)
    const int c = tid;
    if (c > k) {
      const float lck = Lc[c * ND + k];
      for (int r = c; r < ND; ++r)
        Lc[r * ND + c] -= Lc[r * ND + k] * lck;
    }
    __syncthreads();
  }
  if (tid == 0) {
    float s = 0.0f;
    for (int k = 0; k < ND; ++k) s += logf(Lc[k * ND + k]);
    logdet[l] = 2.0f * s;
  }
}

__global__ void inverse_kernel(const float* __restrict__ chol, float* __restrict__ sinv,
                               float* __restrict__ Y) {
  const int l = blockIdx.x, c = threadIdx.x;  // thread c owns column c
  const float* Lc = chol + (size_t)l * ND * ND;
  float* y = Y + ((size_t)l * ND + c) * ND;
  for (int k = 0; k < ND; ++k) {               // forward solve  L y = e_c
    float s = (k == c) ? 1.0f : 0.0f;
    for (int j = 0; j < k; ++j) s -= Lc[k * ND + j] * y[j];
    y[k] = s / Lc[k * ND + k];
  }
  float* Sv = sinv + (size_t)l * ND * ND;
  for (int k = ND - 1; k >= 0; --k) {          // backward solve L^T x = y
    float s = y[k];
    for (int j = k + 1; j < ND; ++j) s -= Lc[j * ND + k] * Sv[(size_t)j * ND + c];
    Sv[(size_t)k * ND + c] = s / Lc[k * ND + k];
  }
}

__global__ void kl_kernel(const float* __restrict__ sig, const float* __restrict__ sinv,
                          const float* __restrict__ mus, const float* __restrict__ logdet,
                          const float* __restrict__ counts, float* __restrict__ klp) {
  const int i = blockIdx.x / NL, j = blockIdx.x % NL, tid = threadIdx.x;
  if (i == j) { if (tid == 0) klp[blockIdx.x] = 0.0f; return; }
  __shared__ float sdiff[ND];
  __shared__ float red[256];
  sdiff[tid] = mus[(size_t)i * ND + tid] - mus[(size_t)j * ND + tid];
  __syncthreads();
  const float* Sj = sinv + (size_t)j * ND * ND;
  const float* Si = sig + (size_t)i * ND * ND;
  float acc = 0.0f;
  for (int e = tid; e < ND * ND; e += 256) {   // tr(Sj^-1 Si); ND is 2^8 -> shifts
    int a = e >> 8, b = e & 255;
    acc += Sj[(size_t)a * ND + b] * Si[(size_t)b * ND + a];
  }
  float wv = 0.0f;                              // (Sj^-1 diff)[tid]
  for (int b = 0; b < ND; ++b) wv += Sj[(size_t)tid * ND + b] * sdiff[b];
  acc += sdiff[tid] * wv;                       // + mahalanobis contribution
  red[tid] = acc; __syncthreads();
  for (int st = 128; st > 0; st >>= 1) {
    if (tid < st) red[tid] += red[tid + st];
    __syncthreads();
  }
  if (tid == 0) {
    float kl = 0.5f * (logdet[j] - logdet[i] - (float)ND + red[0]);
    klp[blockIdx.x] = counts[i] * counts[j] * kl;
  }
}

__global__ void final_kernel(const float* __restrict__ klp, float* __restrict__ out) {
  __shared__ float red[128];
  const int tid = threadIdx.x;
  float s = 0.0f;
  for (int e = tid; e < NL * NL; e += 128) s += klp[e];
  red[tid] = s; __syncthreads();
  for (int st = 64; st > 0; st >>= 1) {
    if (tid < st) red[tid] += red[tid + st];
    __syncthreads();
  }
  if (tid == 0) out[0] = red[0] / 5.49755813888e11f;  // / B^3
}

// ---------------------------------------------------------------------------

extern "C" void kernel_launch(void* const* d_in, const int* in_sizes, int n_in,
                              void* d_out, int out_size, void* d_ws, size_t ws_size,
                              hipStream_t stream) {
  (void)in_sizes; (void)n_in; (void)out_size; (void)ws_size;
  const float* mu  = (const float*)d_in[0];
  const int*   lbl = (const int*)d_in[1];
  float* out = (float*)d_out;

  char* base = (char*)d_ws;
  size_t off = 0;
  auto take = [&](size_t bytes) -> char* {
    char* p = base + off;
    off += (bytes + 255) & ~(size_t)255;
    return p;
  };

  float*  mean   = (float*) take((size_t)NF * 4);
  __bf16* Xcb    = (__bf16*)take((size_t)NB * NF * 2);
  float*  Cf     = (float*) take((size_t)NF * NF * 4);
  __bf16* Cb     = (__bf16*)take((size_t)NF * NF * 2);
  float*  Q      = (float*) take((size_t)NF * ND * 4);
  __bf16* Qb     = (__bf16*)take((size_t)NF * ND * 2);
  float*  Z      = (float*) take((size_t)NF * ND * 4);
  float*  P      = (float*) take((size_t)NB * ND * 4);
  __bf16* Pb     = (__bf16*)take((size_t)NB * ND * 2);
  float*  counts = (float*) take((size_t)NL * 4);
  float*  mus    = (float*) take((size_t)NL * ND * 4);
  float*  sig    = (float*) take((size_t)NL * ND * ND * 4);
  float*  chol   = (float*) take((size_t)NL * ND * ND * 4);
  float*  sinv   = (float*) take((size_t)NL * ND * ND * 4);
  float*  Yscr   = (float*) take((size_t)NL * ND * ND * 4);
  float*  scat   = (float*) take((size_t)NL * KS * ND * ND * 4);  // split-K partials
  float*  logdet = (float*) take((size_t)NL * 4);
  float*  klp    = (float*) take((size_t)NL * NL * 4);

  // 1) center + cast
  mean_kernel<<<NF / 256, 256, 0, stream>>>(mu, mean);
  center_cast_kernel<<<(NB * NF) / 256, 256, 0, stream>>>(mu, mean, Xcb);

  // 2) Gram matrix C = Xc^T Xc  (WMMA bf16, K=8192)
  ata_wmma_kernel<<<dim3(NF / 128, NF / 128, 1), 256, 0, stream>>>(
      Xcb, NB, NF, (const int*)nullptr, 0, Cf);
  cast_f32_bf16_kernel<<<((size_t)NF * NF + 255) / 256, 256, 0, stream>>>(
      Cf, Cb, (size_t)NF * NF);

  // 3) subspace power iteration for top-256 directions
  init_q_kernel<<<(NF * ND) / 256, 256, 0, stream>>>(Q, Qb);
  for (int it = 0; it < 2; ++it) {
    gemm_nn_wmma_kernel<<<dim3(NF / 128, ND / 128), 256, 0, stream>>>(
        Cb, Qb, Z, NF, NF, ND);
    colnorm_kernel<<<ND, 256, 0, stream>>>(Z, Q, Qb);
  }

  // 4) projection P = Xc Q  (WMMA bf16)
  gemm_nn_wmma_kernel<<<dim3(NB / 128, ND / 128), 256, 0, stream>>>(
      Xcb, Qb, P, NB, NF, ND);
  cast_f32_bf16_kernel<<<((size_t)NB * ND + 255) / 256, 256, 0, stream>>>(
      P, Pb, (size_t)NB * ND);

  // 5) per-label stats: counts, means, masked scatter (WMMA, split-K x8)
  counts_kernel<<<1, 256, 0, stream>>>(lbl, counts);
  mus_kernel<<<NL, ND, 0, stream>>>(lbl, P, counts, mus);
  ata_wmma_kernel<<<dim3(ND / 128, ND / 128, NL * KS), 256, 0, stream>>>(
      Pb, NB, ND, lbl, KSHIFT, scat);
  reduce_ksplit_kernel<<<((size_t)NL * ND * ND) / 256, 256, 0, stream>>>(scat, sig);
  sig_kernel<<<((size_t)NL * ND * ND) / 256, 256, 0, stream>>>(sig, counts, mus);

  // 6) Cholesky, inverse, pairwise KL, weighted sum
  cholesky_kernel<<<NL, 256, 0, stream>>>(sig, chol, logdet);
  inverse_kernel<<<NL, ND, 0, stream>>>(chol, sinv, Yscr);
  kl_kernel<<<NL * NL, 256, 0, stream>>>(sig, sinv, mus, logdet, counts, klp);
  final_kernel<<<1, 128, 0, stream>>>(klp, out);
}
